// Qwen3TTSTokenizerDecoderAttention_81930796139166
// MI455X (gfx1250) — matrix operand order repaired and checked by reading, not compiled, not run on hardware
//
#include <hip/hip_runtime.h>
#include <hip/hip_bf16.h>

#define H_MODEL 1024
#define N_HEADS 16
#define N_KV_HEADS 4
#define HEAD_DIM 64
#define SWIN 512
#define BATCH 2
#define SEQ 2048

typedef _Float16 h16_t;
typedef __attribute__((ext_vector_type(8)))  _Float16 v8h;
typedef __attribute__((ext_vector_type(16))) _Float16 v16h;
typedef __attribute__((ext_vector_type(8)))  float    v8f;

__device__ __forceinline__ v16h concat8(v8h lo, v8h hi) {
  return __builtin_shufflevector(lo, hi, 0,1,2,3,4,5,6,7,8,9,10,11,12,13,14,15);
}
__device__ __forceinline__ v8f zero8f() {
  v8f z = {0.f,0.f,0.f,0.f,0.f,0.f,0.f,0.f};
  return z;
}
__device__ __forceinline__ v8f wmma_f16(v16h a, v16h b, v8f c) {
  return __builtin_amdgcn_wmma_f32_16x16x32_f16(false, a, false, b, (short)0, c, false, false);
}

// ---------------- fused conversions (one launch) ----------------
// segments: hs (4194304) | Wq (1048576) | Wk (262144) | Wv (262144) | Wo (1048576)
__global__ __launch_bounds__(256) void convert_all_kernel(
    const float* __restrict__ hs, const float* __restrict__ Wq,
    const float* __restrict__ Wk, const float* __restrict__ Wv,
    const float* __restrict__ Wo,
    h16_t* __restrict__ hs16, h16_t* __restrict__ wqkv16, h16_t* __restrict__ wo16) {
  int i = blockIdx.x * 256 + threadIdx.x;
  if (i < 4194304) { hs16[i] = (h16_t)hs[i]; return; }
  i -= 4194304;
  if (i < 1048576) { wqkv16[i] = (h16_t)Wq[i]; return; }
  i -= 1048576;
  if (i < 262144) { wqkv16[1048576 + i] = (h16_t)Wk[i]; return; }
  i -= 262144;
  if (i < 262144) { wqkv16[1310720 + i] = (h16_t)Wv[i]; return; }
  i -= 262144;
  if (i < 1048576) { wo16[i] = (h16_t)Wo[i]; }
}

// ---------------- fused QKV GEMM + RoPE ----------------
// A = hs16 (4096 x 1024 row major), W = wqkv16 (1536 x 1024 row major, k contiguous)
// One wave -> 32 rows x 64 cols (one head-strip), B fragments reused across both row tiles.
__global__ __launch_bounds__(256) void gemm_qkv_rope_kernel(
    const h16_t* __restrict__ hs16, const h16_t* __restrict__ wqkv,
    const float* __restrict__ cosp, const float* __restrict__ sinp,
    h16_t* __restrict__ q16, h16_t* __restrict__ k16, h16_t* __restrict__ v16T) {
  const int lane = threadIdx.x & 31;
  const int wid  = threadIdx.x >> 5;
  const int task = blockIdx.x * 8 + wid;          // 3072 tasks
  const int NSTRIPS = (N_HEADS + 2 * N_KV_HEADS); // 24 strips of 64
  const int nstrip = task % NSTRIPS;
  const int m0     = (task / NSTRIPS) * 32;       // 0..4064
  const int half = lane >> 4;
  const int ln   = lane & 15;

  const h16_t* arow0 = hs16 + (size_t)(m0 + ln) * H_MODEL;
  const h16_t* arow1 = arow0 + (size_t)16 * H_MODEL;
  const h16_t* wbase = wqkv + (size_t)(nstrip * 64 + ln) * H_MODEL;

  v8f acc[2][4];
#pragma unroll
  for (int ms = 0; ms < 2; ++ms)
#pragma unroll
    for (int t = 0; t < 4; ++t) acc[ms][t] = zero8f();

  for (int kb = 0; kb < H_MODEL; kb += 32) {
    __builtin_prefetch(arow0 + kb + 128, 0, 1);   // global_prefetch next k-steps
    __builtin_prefetch(arow1 + kb + 128, 0, 1);
    v16h af0 = concat8(*(const v8h*)(arow0 + kb + half * 8),
                       *(const v8h*)(arow0 + kb + 16 + half * 8));
    v16h af1 = concat8(*(const v8h*)(arow1 + kb + half * 8),
                       *(const v8h*)(arow1 + kb + 16 + half * 8));
#pragma unroll
    for (int t = 0; t < 4; ++t) {
      const h16_t* bp = wbase + (size_t)(t * 16) * H_MODEL + kb + half * 8;
      v16h bf = concat8(*(const v8h*)bp, *(const v8h*)(bp + 16));
      acc[0][t] = wmma_f16(af0, bf, acc[0][t]);
      acc[1][t] = wmma_f16(af1, bf, acc[1][t]);
    }
  }

#pragma unroll
  for (int ms = 0; ms < 2; ++ms) {
#pragma unroll
    for (int r = 0; r < 8; ++r) {
      const int row = m0 + ms * 16 + half * 8 + r;
      const int bb = row / SEQ;
      const int s  = row % SEQ;
      if (nstrip < 16) {                            // Q head = nstrip, RoPE
        const int hh = nstrip;
#pragma unroll
        for (int t = 0; t < 4; ++t) {
          const int d = t * 16 + ln;
          const float x  = acc[ms][t][r];
          const float pv = acc[ms][t ^ 2][r];       // rotate-half partner (d +/- 32)
          const float c  = cosp[s * 64 + d];
          const float sn = sinp[s * 64 + d];
          const float val = (d < 32) ? (x * c - pv * sn) : (x * c + pv * sn);
          q16[(((size_t)(bb * N_HEADS + hh)) * SEQ + s) * 64 + d] = (h16_t)val;
        }
      } else if (nstrip < 20) {                     // K kv-head, RoPE
        const int kvh = nstrip - 16;
#pragma unroll
        for (int t = 0; t < 4; ++t) {
          const int d = t * 16 + ln;
          const float x  = acc[ms][t][r];
          const float pv = acc[ms][t ^ 2][r];
          const float c  = cosp[s * 64 + d];
          const float sn = sinp[s * 64 + d];
          const float val = (d < 32) ? (x * c - pv * sn) : (x * c + pv * sn);
          k16[(((size_t)(bb * N_KV_HEADS + kvh)) * SEQ + s) * 64 + d] = (h16_t)val;
        }
      } else {                                      // V kv-head, transposed (d-major)
        const int kvh = nstrip - 20;
#pragma unroll
        for (int t = 0; t < 4; ++t) {
          const int d = t * 16 + ln;
          v16T[(((size_t)(bb * N_KV_HEADS + kvh)) * 64 + d) * SEQ + s] = (h16_t)acc[ms][t][r];
        }
      }
    }
  }
}

// ---------------- flash attention (sliding window, GQA) ----------------
__global__ __launch_bounds__(256) void attn_kernel(
    const h16_t* __restrict__ q16, const h16_t* __restrict__ k16,
    const h16_t* __restrict__ v16T, h16_t* __restrict__ attn16) {
  __shared__ _Float16 smem[8][16 * 32];
  const int lane = threadIdx.x & 31;
  const int wid  = threadIdx.x >> 5;
  const int task = blockIdx.x * 8 + wid;            // 4096 tasks
  const int QT = SEQ / 16;
  const int qtile = task % QT;
  const int hh = (task / QT) % N_HEADS;
  const int bb = task / (QT * N_HEADS);
  const int q0 = qtile * 16;
  const int half = lane >> 4;
  const int ln   = lane & 15;
  const int kvh = hh / (N_HEADS / N_KV_HEADS);

  const h16_t* qb = q16 + (((size_t)(bb * N_HEADS + hh)) * SEQ + q0) * 64;
  const h16_t* kp = k16 + ((size_t)(bb * N_KV_HEADS + kvh)) * SEQ * 64;
  const h16_t* vp = v16T + ((size_t)(bb * N_KV_HEADS + kvh)) * 64 * (size_t)SEQ;

  v16h qf[2];
#pragma unroll
  for (int t = 0; t < 2; ++t) {
    const h16_t* p = qb + ln * 64 + t * 32 + half * 8;
    qf[t] = concat8(*(const v8h*)p, *(const v8h*)(p + 16));
  }

  float mrow[8], lrow[8];
  v8f oacc[4];
#pragma unroll
  for (int r = 0; r < 8; ++r) { mrow[r] = -1e30f; lrow[r] = 0.f; }
#pragma unroll
  for (int t = 0; t < 4; ++t) oacc[t] = zero8f();

  int jlo = q0 - (SWIN - 1); if (jlo < 0) jlo = 0;
  const int kb0 = jlo & ~31;

  for (int kb = kb0; kb <= q0 + 15; kb += 32) {
    v8f sc[2];
#pragma unroll
    for (int st = 0; st < 2; ++st) {
      v8f a = zero8f();
#pragma unroll
      for (int t2 = 0; t2 < 2; ++t2) {
        const h16_t* p = kp + (size_t)(kb + st * 16 + ln) * 64 + t2 * 32 + half * 8;
        v16h bf = concat8(*(const v8h*)p, *(const v8h*)(p + 16));
        a = wmma_f16(qf[t2], bf, a);
      }
      sc[st] = a;
    }
    // scale + analytic causal/window mask
#pragma unroll
    for (int st = 0; st < 2; ++st)
#pragma unroll
      for (int r = 0; r < 8; ++r) {
        const int i = q0 + half * 8 + r;
        const int j = kb + st * 16 + ln;
        const bool ok = (j <= i) && (i - j < SWIN);
        sc[st][r] = sc[st][r] * 0.125f + (ok ? 0.f : -1e9f);
      }
    // online softmax (16-lane reductions stay within fragment half on wave32)
#pragma unroll
    for (int r = 0; r < 8; ++r) {
      float rm = fmaxf(sc[0][r], sc[1][r]);
#pragma unroll
      for (int msk = 1; msk < 16; msk <<= 1) rm = fmaxf(rm, __shfl_xor(rm, msk, 32));
      const float mnew = fmaxf(mrow[r], rm);
      const float alpha = __expf(mrow[r] - mnew);
      mrow[r] = mnew;
      const float p0 = __expf(sc[0][r] - mnew);
      const float p1 = __expf(sc[1][r] - mnew);
      float rs = p0 + p1;
#pragma unroll
      for (int msk = 1; msk < 16; msk <<= 1) rs += __shfl_xor(rs, msk, 32);
      lrow[r] = lrow[r] * alpha + rs;
#pragma unroll
      for (int t = 0; t < 4; ++t) oacc[t][r] *= alpha;
      const int m = half * 8 + r;
      smem[wid][m * 32 + ln]      = (h16_t)p0;
      smem[wid][m * 32 + 16 + ln] = (h16_t)p1;
    }
    asm volatile("s_wait_dscnt 0" ::: "memory");   // D-layout P -> A-layout via LDS
    v16h pf;
    {
      const _Float16* p = &smem[wid][ln * 32 + half * 8];
      pf = concat8(*(const v8h*)p, *(const v8h*)(p + 16));
    }
#pragma unroll
    for (int t = 0; t < 4; ++t) {
      const h16_t* p = vp + (size_t)(t * 16 + ln) * SEQ + kb + half * 8;
      v16h bf = concat8(*(const v8h*)p, *(const v8h*)(p + 16));
      oacc[t] = wmma_f16(pf, bf, oacc[t]);
    }
    asm volatile("s_wait_dscnt 0" ::: "memory");   // LDS reads drain before next block's stores
  }

#pragma unroll
  for (int r = 0; r < 8; ++r) {
    const float inv = 1.0f / lrow[r];
    const int s = q0 + half * 8 + r;
#pragma unroll
    for (int t = 0; t < 4; ++t) {
      attn16[((size_t)(bb * SEQ + s)) * (N_HEADS * HEAD_DIM) + hh * 64 + t * 16 + ln] =
          (h16_t)(oacc[t][r] * inv);
    }
  }
}

// ---------------- output projection (32x64 tile per wave) ----------------
__global__ __launch_bounds__(256) void gemm_out_kernel(
    const h16_t* __restrict__ attn16, const h16_t* __restrict__ wo16,
    float* __restrict__ out) {
  const int lane = threadIdx.x & 31;
  const int wid  = threadIdx.x >> 5;
  const int task = blockIdx.x * 8 + wid;           // 2048 tasks
  const int NSTRIPS = H_MODEL / 64;                // 16
  const int nstrip = task % NSTRIPS;
  const int m0 = (task / NSTRIPS) * 32;
  const int half = lane >> 4;
  const int ln = lane & 15;

  const h16_t* arow0 = attn16 + (size_t)(m0 + ln) * H_MODEL;
  const h16_t* arow1 = arow0 + (size_t)16 * H_MODEL;
  const h16_t* wbase = wo16 + (size_t)(nstrip * 64 + ln) * H_MODEL;

  v8f acc[2][4];
#pragma unroll
  for (int ms = 0; ms < 2; ++ms)
#pragma unroll
    for (int t = 0; t < 4; ++t) acc[ms][t] = zero8f();

  for (int kb = 0; kb < H_MODEL; kb += 32) {
    __builtin_prefetch(arow0 + kb + 128, 0, 1);
    __builtin_prefetch(arow1 + kb + 128, 0, 1);
    v16h af0 = concat8(*(const v8h*)(arow0 + kb + half * 8),
                       *(const v8h*)(arow0 + kb + 16 + half * 8));
    v16h af1 = concat8(*(const v8h*)(arow1 + kb + half * 8),
                       *(const v8h*)(arow1 + kb + 16 + half * 8));
#pragma unroll
    for (int t = 0; t < 4; ++t) {
      const h16_t* bp = wbase + (size_t)(t * 16) * H_MODEL + kb + half * 8;
      v16h bf = concat8(*(const v8h*)bp, *(const v8h*)(bp + 16));
      acc[0][t] = wmma_f16(af0, bf, acc[0][t]);
      acc[1][t] = wmma_f16(af1, bf, acc[1][t]);
    }
  }
#pragma unroll
  for (int ms = 0; ms < 2; ++ms)
#pragma unroll
    for (int r = 0; r < 8; ++r) {
      const int row = m0 + ms * 16 + half * 8 + r;
#pragma unroll
      for (int t = 0; t < 4; ++t)
        out[(size_t)row * H_MODEL + nstrip * 64 + t * 16 + ln] = acc[ms][t][r];
    }
}

extern "C" void kernel_launch(void* const* d_in, const int* in_sizes, int n_in,
                              void* d_out, int out_size, void* d_ws, size_t ws_size,
                              hipStream_t stream) {
  const float* hs   = (const float*)d_in[0];
  const float* cosp = (const float*)d_in[1];
  const float* sinp = (const float*)d_in[2];
  // d_in[3] = attention_mask (unused; mask computed analytically)
  const float* Wq = (const float*)d_in[4];
  const float* Wk = (const float*)d_in[5];
  const float* Wv = (const float*)d_in[6];
  const float* Wo = (const float*)d_in[7];
  float* out = (float*)d_out;

  char* ws = (char*)d_ws;
  h16_t* hs16   = (h16_t*)(ws);                          // 8,388,608 B
  h16_t* wqkv16 = (h16_t*)(ws + 8388608);                // 3,145,728 B
  h16_t* wo16   = (h16_t*)(ws + 11534336);               // 2,097,152 B
  h16_t* q16    = (h16_t*)(ws + 13631488);               // 8,388,608 B
  h16_t* k16    = (h16_t*)(ws + 22020096);               // 2,097,152 B
  h16_t* v16T   = (h16_t*)(ws + 24117248);               // 2,097,152 B
  h16_t* attn16 = (h16_t*)(ws + 26214400);               // 8,388,608 B  (total ~33 MB)

  const int ncvt = 4194304 + 1048576 + 262144 + 262144 + 1048576;  // 6,815,744
  convert_all_kernel<<<(ncvt + 255) / 256, 256, 0, stream>>>(
      hs, Wq, Wk, Wv, Wo, hs16, wqkv16, wo16);

  gemm_qkv_rope_kernel<<<384, 256, 0, stream>>>(hs16, wqkv16, cosp, sinp, q16, k16, v16T);
  attn_kernel<<<512, 256, 0, stream>>>(q16, k16, v16T, attn16);
  gemm_out_kernel<<<256, 256, 0, stream>>>(attn16, wo16, out);
}
